// Decoder_51531017617909
// MI455X (gfx1250) — matrix operand order repaired and checked by reading, not compiled, not run on hardware
//
#include <hip/hip_runtime.h>
#include <math.h>

#define B_    64
#define TK_   800
#define TQ_   64
#define H_    512
#define H2_   1024
#define E_    256
#define V_    50000
#define OOV_  50
#define VO_   (V_ + OOV_)

typedef __attribute__((ext_vector_type(8)))  float  v8f;
typedef __attribute__((ext_vector_type(4)))  float  v4f;
typedef __attribute__((ext_vector_type(16))) __bf16 v16bf;

union BF16x16 { unsigned short u[16]; v16bf v; };

static __device__ __forceinline__ unsigned short f2bf(float f) {
    union { float f; unsigned int u; } x; x.f = f;
    unsigned int u = x.u;
    unsigned int r = u + 0x7fffu + ((u >> 16) & 1u);   // round-to-nearest-even
    return (unsigned short)(r >> 16);
}

static __device__ __forceinline__ float sigmoidf_(float x) {
    return 1.0f / (1.0f + expf(-x));
}

// ---------------------------------------------------------------- f32 -> bf16
__global__ void k_f2bf(const float* __restrict__ src, unsigned short* __restrict__ dst, int n) {
    int i = blockIdx.x * 256 + threadIdx.x;
    if (i < n) dst[i] = f2bf(src[i]);
}

// ------------------------------------------------- embedding + xc projection
// x[b,e] = xc_b[e] + sum_k xc_W[e,k] * cat(c_t_1[b], emb_W[y[b]])[k]
// b in low bits -> 64 lanes broadcast one weight row.
__global__ __launch_bounds__(256) void k_x(const int* __restrict__ y,
                                           const float* __restrict__ ct1,
                                           const float* __restrict__ embW,
                                           const float* __restrict__ xcW,
                                           const float* __restrict__ xcb,
                                           float* __restrict__ x) {
    int idx = blockIdx.x * 256 + threadIdx.x;   // B*E
    int b = idx & 63, e = idx >> 6;
    const float* wr = xcW + (size_t)e * (H2_ + E_);
    const float* c  = ct1 + (size_t)b * H2_;
    float acc = xcb[e];
    for (int k = 0; k < H2_; ++k) acc += wr[k] * c[k];
    const float* em = embW + (size_t)y[b] * E_;
    for (int k = 0; k < E_; ++k) acc += wr[H2_ + k] * em[k];
    x[b * E_ + e] = acc;
}

// ---------------------------------------------------------------- LSTM gates
__global__ __launch_bounds__(256) void k_gates(const float* __restrict__ x,
                                               const float* __restrict__ h0,
                                               const float* __restrict__ Wih,
                                               const float* __restrict__ Whh,
                                               const float* __restrict__ bih,
                                               const float* __restrict__ bhh,
                                               float* __restrict__ gates) {
    int idx = blockIdx.x * 256 + threadIdx.x;   // B * 4H = 131072
    int b = idx & 63, j = idx >> 6;
    float acc = bih[j] + bhh[j];
    const float* xr = x + (size_t)b * E_;
    const float* wi = Wih + (size_t)j * E_;
    for (int k = 0; k < E_; ++k) acc += wi[k] * xr[k];
    const float* hr = h0 + (size_t)b * H_;
    const float* wh = Whh + (size_t)j * H_;
    for (int k = 0; k < H_; ++k) acc += wh[k] * hr[k];
    gates[(size_t)b * (4 * H_) + j] = acc;
}

// ------------------------------------------------------------- LSTM nonlinear
__global__ __launch_bounds__(512) void k_lstm(const float* __restrict__ gates,
                                              const float* __restrict__ c0,
                                              float* __restrict__ shat,
                                              float* __restrict__ hout,
                                              float* __restrict__ cout) {
    int b = blockIdx.x, h = threadIdx.x;
    const float* g = gates + (size_t)b * 2048;
    float gi = g[h], gf = g[H_ + h], gg = g[2 * H_ + h], go = g[3 * H_ + h];
    float cn = sigmoidf_(gf) * c0[b * H_ + h] + sigmoidf_(gi) * tanhf(gg);
    float hn = sigmoidf_(go) * tanhf(cn);
    shat[b * H2_ + h]       = hn;
    shat[b * H2_ + H_ + h]  = cn;
    hout[b * H_ + h] = hn;
    cout[b * H_ + h] = cn;
}

// ----------------------------------------------- generic dst = src @ W.T + b
__global__ __launch_bounds__(256) void k_gemv(const float* __restrict__ src,
                                              const float* __restrict__ W,
                                              const float* __restrict__ bias,
                                              float* __restrict__ dst,
                                              int N, int K) {
    int idx = blockIdx.x * 256 + threadIdx.x;   // B * N
    int b = idx & 63, n = idx >> 6;
    const float* wr = W + (size_t)n * K;
    const float* s  = src + (size_t)b * K;
    float acc = bias ? bias[n] : 0.0f;
    for (int k = 0; k < K; ++k) acc += wr[k] * s[k];
    dst[(size_t)b * N + n] = acc;
}

// ------------------------------------------------------ fused WMMA scorer ---
// scores[row] = sum_n v[n] * tanh( (henc[row] @ Wh.T)[n] + dec[b,n]
//                                   (+ qfeat[b,n]) (+ cov[row]*Wc[n]) )
// bf16 inputs, f32 accumulate. 32 rows per workgroup, 8 waves x 128 N-cols.
// A tile (32x32) staged cooperatively in double-buffered LDS (bf16) so the
// f32->bf16 conversion + global A loads happen once per block, not per wave.
__global__ __launch_bounds__(256) void k_score(const float* __restrict__ henc, int T,
                                               const unsigned short* __restrict__ whbf,
                                               const float* __restrict__ dec,
                                               const float* __restrict__ qfeat,   // nullable
                                               const float* __restrict__ cov,     // nullable [B*T]
                                               const float* __restrict__ wc,      // used iff cov
                                               const float* __restrict__ v,
                                               float* __restrict__ scores) {
    __shared__ unsigned short Abf[2][32][40];   // 40-ushort row stride: bank-friendly
    __shared__ float red[32][128];
    const int tid  = threadIdx.x;
    const int wave = tid >> 5;
    const int lane = tid & 31;
    const int l15  = lane & 15;
    const int hi   = lane >> 4;           // 0 | 1
    const int row0 = blockIdx.x * 32;
    const int b    = row0 / T;
    const int n0   = wave * 128;

    // staging assignment: thread -> (row, 4-wide k chunk)
    const int srow = tid >> 3;            // 0..31
    const int scol = (tid & 7) * 4;       // 0,4,...,28
    const float* sbase = henc + (size_t)(row0 + srow) * H2_ + scol;

    // stage K-step 0 into buffer 0
    {
        v4f a = *(const v4f*)(sbase);
        unsigned int w0 = (unsigned int)f2bf(a.x) | ((unsigned int)f2bf(a.y) << 16);
        unsigned int w1 = (unsigned int)f2bf(a.z) | ((unsigned int)f2bf(a.w) << 16);
        *(unsigned int*)&Abf[0][srow][scol]     = w0;
        *(unsigned int*)&Abf[0][srow][scol + 2] = w1;
    }

    v8f acc[2][8] = {};

    for (int kk = 0; kk < H2_; kk += 32) {
        const int p = (kk >> 5) & 1;
        __syncthreads();                  // buffer p staged; prior reads of p^1 done
        if (kk + 32 < H2_) {              // prefetch next K-step into buffer p^1
            v4f a = *(const v4f*)(sbase + kk + 32);
            unsigned int w0 = (unsigned int)f2bf(a.x) | ((unsigned int)f2bf(a.y) << 16);
            unsigned int w1 = (unsigned int)f2bf(a.z) | ((unsigned int)f2bf(a.w) << 16);
            *(unsigned int*)&Abf[p ^ 1][srow][scol]     = w0;
            *(unsigned int*)&Abf[p ^ 1][srow][scol + 2] = w1;
        }
        // A fragments from LDS: contiguous k order, same as B
        // (lane<16: K 0-15 / lane>=16: K 16-31)
        v16bf afrag[2];
        #pragma unroll
        for (int mt = 0; mt < 2; ++mt) {
            const unsigned short* lp = &Abf[p][mt * 16 + l15][hi * 16];
            BF16x16 t;
            __builtin_memcpy(&t.u[0], lp, 16);       // ds_load_b128
            __builtin_memcpy(&t.u[8], lp + 8, 16);   // ds_load_b128
            afrag[mt] = t.v;
        }
        // B fragments: contiguous k order, same as A -> lane reads 16 consecutive K of row n
        #pragma unroll
        for (int s = 0; s < 8; ++s) {
            const int n = n0 + s * 16 + l15;
            const v16bf bfrag = *(const v16bf*)(whbf + (size_t)n * H2_ + kk + hi * 16);
            acc[0][s] = __builtin_amdgcn_wmma_f32_16x16x32_bf16(
                false, afrag[0], false, bfrag, (short)0, acc[0][s], false, false);
            acc[1][s] = __builtin_amdgcn_wmma_f32_16x16x32_bf16(
                false, afrag[1], false, bfrag, (short)0, acc[1][s], false, false);
        }
    }

    // epilogue: + dec (+qfeat) (+cov*Wc), tanh, *v[n], partial reduce per lane
    #pragma unroll
    for (int mt = 0; mt < 2; ++mt) {
        #pragma unroll
        for (int r = 0; r < 8; ++r) {
            const int M   = mt * 16 + r + hi * 8;
            const int row = row0 + M;
            const float covterm = cov ? cov[row] : 0.0f;
            float p = 0.0f;
            #pragma unroll
            for (int s = 0; s < 8; ++s) {
                const int n = n0 + s * 16 + l15;
                float f = acc[mt][s][r] + dec[b * H2_ + n];
                if (qfeat) f += qfeat[b * H2_ + n];
                if (cov)   f += covterm * wc[n];
                p += tanhf(f) * v[n];
            }
            red[M][wave * 16 + l15] = p;
        }
    }
    __syncthreads();
    if (tid < 32) {                           // fixed-order (deterministic) row sum
        float s = 0.0f;
        for (int j = 0; j < 128; ++j) s += red[tid][j];
        scores[row0 + tid] = s;
    }
}

// ---------------------- query softmax (mask-renorm) + query context vector --
__global__ __launch_bounds__(256) void k_qsoft(const float* __restrict__ scores,
                                               const float* __restrict__ mask,
                                               const float* __restrict__ qenc,
                                               float* __restrict__ qct) {
    __shared__ float a[TQ_];
    __shared__ float r[256];
    int b = blockIdx.x, tid = threadIdx.x;
    float m = -1e30f;
    if (tid < TQ_) m = scores[b * TQ_ + tid];
    r[tid] = m; __syncthreads();
    for (int s = 128; s > 0; s >>= 1) { if (tid < s) r[tid] = fmaxf(r[tid], r[tid + s]); __syncthreads(); }
    float mx = r[0]; __syncthreads();
    float e = 0.0f;
    if (tid < TQ_) { e = expf(scores[b * TQ_ + tid] - mx) * mask[b * TQ_ + tid]; a[tid] = e; }
    r[tid] = e; __syncthreads();
    for (int s = 128; s > 0; s >>= 1) { if (tid < s) r[tid] += r[tid + s]; __syncthreads(); }
    float inv = 1.0f / r[0]; __syncthreads();
    if (tid < TQ_) a[tid] *= inv;
    __syncthreads();
    for (int n = tid; n < H2_; n += 256) {
        const float* hp = qenc + (size_t)b * TQ_ * H2_ + n;
        float acc = 0.0f;
        for (int t = 0; t < TQ_; ++t) acc += a[t] * hp[(size_t)t * H2_];
        qct[b * H2_ + n] = acc;
    }
}

// -------- main softmax + coverage update + context vector (writes outputs) --
__global__ __launch_bounds__(256) void k_mainsoft(const float* __restrict__ scores,
                                                  const float* __restrict__ mask,
                                                  const float* __restrict__ cov_in,
                                                  const float* __restrict__ enc,
                                                  float* __restrict__ attn_out,
                                                  float* __restrict__ cov_out,
                                                  float* __restrict__ ct) {
    __shared__ float a[TK_];
    __shared__ float r[256];
    int b = blockIdx.x, tid = threadIdx.x;
    float m = -1e30f;
    for (int t = tid; t < TK_; t += 256) m = fmaxf(m, scores[b * TK_ + t]);
    r[tid] = m; __syncthreads();
    for (int s = 128; s > 0; s >>= 1) { if (tid < s) r[tid] = fmaxf(r[tid], r[tid + s]); __syncthreads(); }
    float mx = r[0]; __syncthreads();
    float lsum = 0.0f;
    for (int t = tid; t < TK_; t += 256) {
        float e = expf(scores[b * TK_ + t] - mx) * mask[b * TK_ + t];
        a[t] = e; lsum += e;
    }
    r[tid] = lsum; __syncthreads();
    for (int s = 128; s > 0; s >>= 1) { if (tid < s) r[tid] += r[tid + s]; __syncthreads(); }
    float inv = 1.0f / r[0]; __syncthreads();
    for (int t = tid; t < TK_; t += 256) {
        float av = a[t] * inv; a[t] = av;
        attn_out[b * TK_ + t] = av;
        cov_out[b * TK_ + t]  = cov_in[b * TK_ + t] + av;
    }
    __syncthreads();
    for (int n = tid; n < H2_; n += 256) {
        const float* hp = enc + (size_t)b * TK_ * H2_ + n;
        float acc = 0.0f;
        for (int t = 0; t < TK_; ++t) acc += a[t] * hp[(size_t)t * H2_];
        ct[b * H2_ + n] = acc;
    }
}

// ------------------------------------------------------------------- p_gen --
__global__ __launch_bounds__(256) void k_pgen(const float* __restrict__ ct,
                                              const float* __restrict__ shat,
                                              const float* __restrict__ x,
                                              const float* __restrict__ W,
                                              const float* __restrict__ bias,
                                              float* __restrict__ pgen) {
    __shared__ float r[256];
    int b = blockIdx.x, tid = threadIdx.x;
    float acc = 0.0f;
    for (int j = tid; j < 2304; j += 256) {
        float vj = (j < H2_) ? ct[b * H2_ + j]
                 : (j < 2 * H2_) ? shat[b * H2_ + (j - H2_)]
                 : x[b * E_ + (j - 2 * H2_)];
        acc += W[j] * vj;
    }
    r[tid] = acc; __syncthreads();
    for (int s = 128; s > 0; s >>= 1) { if (tid < s) r[tid] += r[tid + s]; __syncthreads(); }
    if (tid == 0) pgen[b] = sigmoidf_(r[0] + bias[0]);
}

// ---------------------------------------------------------------- out1 -----
__global__ __launch_bounds__(256) void k_out1(const float* __restrict__ shat,
                                              const float* __restrict__ ct,
                                              const float* __restrict__ W,
                                              const float* __restrict__ bias,
                                              float* __restrict__ outv) {
    int idx = blockIdx.x * 256 + threadIdx.x;   // B*H = 32768
    int b = idx & 63, h = idx >> 6;
    const float* wr = W + (size_t)h * (3 * H_);
    const float* hn = shat + (size_t)b * H2_;   // h_new = shat[:, 0:H]
    float acc = bias[h];
    for (int k = 0; k < H_; ++k)  acc += wr[k] * hn[k];
    const float* c = ct + (size_t)b * H2_;
    for (int k = 0; k < H2_; ++k) acc += wr[H_ + k] * c[k];
    outv[(size_t)b * H_ + h] = acc;
}

// ---------------------------------------------------- logits (memory-bound) --
__global__ __launch_bounds__(256) void k_logits(const float* __restrict__ outv,
                                                const float* __restrict__ W,
                                                const float* __restrict__ bias,
                                                float* __restrict__ logits) {
    int idx = blockIdx.x * 256 + threadIdx.x;   // V*B = 3.2M
    int b = idx & 63, vv = idx >> 6;
    const float* wr = W + (size_t)vv * H_;      // broadcast across 64 b-lanes
    const float* s  = outv + (size_t)b * H_;
    float acc = bias[vv];
    for (int k = 0; k < H_; ++k) acc += wr[k] * s[k];
    logits[(size_t)b * V_ + vv] = acc;
}

// ------------------------------------------------- vocab softmax statistics --
__global__ __launch_bounds__(256) void k_vmaxsum(const float* __restrict__ logits,
                                                 float* __restrict__ rmax,
                                                 float* __restrict__ rsum) {
    __shared__ float r[256];
    int b = blockIdx.x, tid = threadIdx.x;
    const float* lp = logits + (size_t)b * V_;
    float m = -1e30f;
    for (int j = tid; j < V_; j += 256) m = fmaxf(m, lp[j]);
    r[tid] = m; __syncthreads();
    for (int s = 128; s > 0; s >>= 1) { if (tid < s) r[tid] = fmaxf(r[tid], r[tid + s]); __syncthreads(); }
    float mx = r[0]; __syncthreads();
    float sum = 0.0f;
    for (int j = tid; j < V_; j += 256) sum += expf(lp[j] - mx);
    r[tid] = sum; __syncthreads();
    for (int s = 128; s > 0; s >>= 1) { if (tid < s) r[tid] += r[tid + s]; __syncthreads(); }
    if (tid == 0) { rmax[b] = mx; rsum[b] = r[0]; }
}

// ---------------------------- final_dist base: p_gen * softmax ++ extra_zeros
__global__ __launch_bounds__(256) void k_final_base(const float* __restrict__ logits,
                                                    const float* __restrict__ rmax,
                                                    const float* __restrict__ rsum,
                                                    const float* __restrict__ pgen,
                                                    const float* __restrict__ xzeros,
                                                    float* __restrict__ fin) {
    long idx = (long)blockIdx.x * 256 + threadIdx.x;
    if (idx >= (long)B_ * VO_) return;
    int b = (int)(idx / VO_);
    int j = (int)(idx % VO_);
    float out;
    if (j < V_) out = pgen[b] * expf(logits[(size_t)b * V_ + j] - rmax[b]) / rsum[b];
    else        out = xzeros[b * OOV_ + (j - V_)];
    fin[idx] = out;
}

// -------- scatter (1-p_gen)*attn into final dist; 1 thread per batch row -----
// Per-thread non-returning atomic adds to the same address stay ordered, and
// rows are disjoint across threads -> deterministic.
__global__ void k_scatter(const int* __restrict__ ebev,
                          const float* __restrict__ attn,
                          const float* __restrict__ pgen,
                          float* __restrict__ fin) {
    int b = threadIdx.x;
    float w = 1.0f - pgen[b];
    float* row = fin + (size_t)b * VO_;
    const int* ix = ebev + (size_t)b * TK_;
    const float* at = attn + (size_t)b * TK_;
    for (int t = 0; t < TK_; ++t) atomicAdd(&row[ix[t]], w * at[t]);
}

// ============================================================================
extern "C" void kernel_launch(void* const* d_in, const int* in_sizes, int n_in,
                              void* d_out, int out_size, void* d_ws, size_t ws_size,
                              hipStream_t stream) {
    const int*   y      = (const int*)  d_in[0];
    const float* h0     = (const float*)d_in[1];
    const float* c0     = (const float*)d_in[2];
    const float* enc    = (const float*)d_in[3];
    const float* qenc   = (const float*)d_in[4];
    const float* emask  = (const float*)d_in[5];
    const float* qmask  = (const float*)d_in[6];
    const float* ct1    = (const float*)d_in[7];
    const float* xzeros = (const float*)d_in[8];
    const int*   ebev   = (const int*)  d_in[9];
    const float* cov    = (const float*)d_in[10];
    const float* embW   = (const float*)d_in[11];
    const float* xcW    = (const float*)d_in[12];
    const float* xcb    = (const float*)d_in[13];
    const float* Wih    = (const float*)d_in[14];
    const float* Whh    = (const float*)d_in[15];
    const float* bih    = (const float*)d_in[16];
    const float* bhh    = (const float*)d_in[17];
    const float* aWh    = (const float*)d_in[18];
    const float* aWq    = (const float*)d_in[19];
    const float* aWc    = (const float*)d_in[20];
    const float* adecW  = (const float*)d_in[21];
    const float* adecb  = (const float*)d_in[22];
    const float* av     = (const float*)d_in[23];
    const float* qWh    = (const float*)d_in[24];
    const float* qdecW  = (const float*)d_in[25];
    const float* qdecb  = (const float*)d_in[26];
    const float* qv     = (const float*)d_in[27];
    const float* pgW    = (const float*)d_in[28];
    const float* pgb    = (const float*)d_in[29];
    const float* o1W    = (const float*)d_in[30];
    const float* o1b    = (const float*)d_in[31];
    const float* o2W    = (const float*)d_in[32];
    const float* o2b    = (const float*)d_in[33];

    // output layout (flat, return order)
    float* fin      = (float*)d_out;                         // [B, V+OOV]
    float* o_h      = fin   + (size_t)B_ * VO_;              // [B, H]
    float* o_c      = o_h   + (size_t)B_ * H_;               // [B, H]
    float* o_ct     = o_c   + (size_t)B_ * H_;               // [B, H2]
    float* o_attn   = o_ct  + (size_t)B_ * H2_;              // [B, TK]
    float* o_pg     = o_attn+ (size_t)B_ * TK_;              // [B]
    float* o_cov    = o_pg  + B_;                            // [B, TK]

    // workspace layout
    char* w = (char*)d_ws;
    unsigned short* whbf  = (unsigned short*)(w);                      // 2 MB
    unsigned short* qwhbf = (unsigned short*)(w + (2u<<20));           // 2 MB
    float* xbuf   = (float*)(w + (4u<<20));                            // 64 KB
    float* gates  = xbuf   + (size_t)B_ * E_;                          // 512 KB
    float* shat   = gates  + (size_t)B_ * 4 * H_;                      // 256 KB
    float* decm   = shat   + (size_t)B_ * H2_;                         // 256 KB
    float* decq   = decm   + (size_t)B_ * H2_;                         // 256 KB
    float* qscore = decq   + (size_t)B_ * H2_;                         // 16 KB
    float* qct    = qscore + (size_t)B_ * TQ_;                         // 256 KB
    float* qfeat  = qct    + (size_t)B_ * H2_;                         // 256 KB
    float* mscore = qfeat  + (size_t)B_ * H2_;                         // 200 KB
    float* outv   = mscore + (size_t)B_ * TK_;                         // 128 KB
    float* logits = outv   + (size_t)B_ * H_;                          // 12.8 MB
    float* rmax   = logits + (size_t)B_ * V_;
    float* rsum   = rmax   + B_;

    // 1) bf16 weight copies for WMMA path
    k_f2bf<<<(H2_*H2_)/256, 256, 0, stream>>>(aWh, whbf, H2_*H2_);
    k_f2bf<<<(H2_*H2_)/256, 256, 0, stream>>>(qWh, qwhbf, H2_*H2_);

    // 2) embedding + xc projection
    k_x<<<(B_*E_)/256, 256, 0, stream>>>(y, ct1, embW, xcW, xcb, xbuf);

    // 3) LSTM
    k_gates<<<(B_*4*H_)/256, 256, 0, stream>>>(xbuf, h0, Wih, Whh, bih, bhh, gates);
    k_lstm<<<B_, H_, 0, stream>>>(gates, c0, shat, o_h, o_c);

    // 4) decoder-feature projections
    k_gemv<<<(B_*H2_)/256, 256, 0, stream>>>(shat, adecW, adecb, decm, H2_, H2_);
    k_gemv<<<(B_*H2_)/256, 256, 0, stream>>>(shat, qdecW, qdecb, decq, H2_, H2_);

    // 5) query attention: fused bf16-WMMA scoring, softmax, context, Wq proj
    k_score<<<(B_*TQ_)/32, 256, 0, stream>>>(qenc, TQ_, qwhbf, decq,
                                             nullptr, nullptr, nullptr, qv, qscore);
    k_qsoft<<<B_, 256, 0, stream>>>(qscore, qmask, qenc, qct);
    k_gemv<<<(B_*H2_)/256, 256, 0, stream>>>(qct, aWq, nullptr, qfeat, H2_, H2_);

    // 6) main attention: fused bf16-WMMA scoring (+qfeat +coverage), softmax,
    //    coverage update, context vector
    k_score<<<(B_*TK_)/32, 256, 0, stream>>>(enc, TK_, whbf, decm,
                                             qfeat, cov, aWc, av, mscore);
    k_mainsoft<<<B_, 256, 0, stream>>>(mscore, emask, cov, enc, o_attn, o_cov, o_ct);

    // 7) p_gen, out1, logits, vocab softmax, final distribution
    k_pgen<<<B_, 256, 0, stream>>>(o_ct, shat, xbuf, pgW, pgb, o_pg);
    k_out1<<<(B_*H_)/256, 256, 0, stream>>>(shat, o_ct, o1W, o1b, outv);
    k_logits<<<(B_*V_)/256, 256, 0, stream>>>(outv, o2W, o2b, logits);
    k_vmaxsum<<<B_, 256, 0, stream>>>(logits, rmax, rsum);
    k_final_base<<<((long)B_*VO_ + 255)/256, 256, 0, stream>>>(logits, rmax, rsum,
                                                               o_pg, xzeros, fin);
    k_scatter<<<1, B_, 0, stream>>>(ebev, o_attn, o_pg, fin);
}
